// LegacySeqFirstSelfAttention_6073083756830
// MI455X (gfx1250) — matrix operand order repaired and checked by reading, not compiled, not run on hardware
//
#include <hip/hip_runtime.h>
#include <cstddef>
#include <cstdint>

typedef __attribute__((ext_vector_type(16))) __bf16 v16bf;
typedef __attribute__((ext_vector_type(8)))  __bf16 v8bf;
typedef __attribute__((ext_vector_type(4)))  __bf16 v4bf;
typedef __attribute__((ext_vector_type(8)))  float  v8f;

#define S_LEN 2048
#define BATCH 2
#define HID   1024
#define NHEAD 16
#define HDIM  64
#define ROWS  (S_LEN * BATCH)   // 4096
#define NCOL  (3 * HID)         // 3072

// scale = HD^-0.5 * S^-0.5 = 0.125 / sqrt(2048)
static constexpr float kScale = 0.0027621358640099515f;

// LDS tile geometry (bf16, row stride 40 elems = 80 B)
#define TILE_ELEMS   (128 * 40)          // 5120 elems = 10240 B
#define TILE_BYTES   (TILE_ELEMS * 2)    // 10240
#define BUF_ELEMS    (4 * TILE_ELEMS)    // sAh,sAl,sBh,sBl
#define BUF_BYTES    (4 * TILE_BYTES)    // 40960

// ---------------------------------------------------------------------------
// Kernel 0: split fp32 -> (bf16 hi, bf16 lo) once, outside the GEMM hot loop.
// x ~= hi + lo ; hi = RN_bf16(x), lo = RN_bf16(x - hi).
// ---------------------------------------------------------------------------
__global__ __launch_bounds__(256) void split_bf16_kernel(
    const float* __restrict__ src, __bf16* __restrict__ hi,
    __bf16* __restrict__ lo, int n4)
{
  const int i = blockIdx.x * 256 + threadIdx.x;
  if (i >= n4) return;
  const float4 x = ((const float4*)src)[i];
  const __bf16 h0 = (__bf16)x.x, h1 = (__bf16)x.y,
               h2 = (__bf16)x.z, h3 = (__bf16)x.w;
  v4bf hv = {h0, h1, h2, h3};
  v4bf lv = {(__bf16)(x.x - (float)h0), (__bf16)(x.y - (float)h1),
             (__bf16)(x.z - (float)h2), (__bf16)(x.w - (float)h3)};
  ((v4bf*)hi)[i] = hv;
  ((v4bf*)lo)[i] = lv;
}

// ---------------------------------------------------------------------------
// Kernel 1: qkv = hidden @ W^T + b   [4096 x 3072], K = 1024 (25.8 GFLOP)
// bf16x3 split-precision WMMA GEMM on pre-split hi/lo operands.
// Double-buffered LDS filled with GLOBAL_LOAD_ASYNC_TO_LDS_B128 (ASYNCcnt),
// next k-step's tiles stream in concurrently with the current step's WMMAs.
// Single barrier per k-step (double buffering makes the trailing one redundant).
// Block: 256 threads = 8 waves, tile 128(M) x 128(N), K-step 32.
// Wave (w&3,w>>2): 32x64 region = 2(M) x 4(N) tiles of 16x16.
// ---------------------------------------------------------------------------
__global__ __launch_bounds__(256) void qkv_gemm_kernel(
    const __bf16* __restrict__ Ah, const __bf16* __restrict__ Al,
    const __bf16* __restrict__ Wh, const __bf16* __restrict__ Wl,
    const float* __restrict__ bias, float* __restrict__ C)
{
  __shared__ __bf16 smem[2 * BUF_ELEMS];   // 80 KB: 2 buffers x 4 tiles

  const int tid  = threadIdx.x;
  const int lane = tid & 31;
  const int wave = tid >> 5;
  const int half = lane >> 4;   // 0/1 lane group
  const int l16  = lane & 15;
  const int r0 = blockIdx.x * 128;
  const int c0 = blockIdx.y * 128;
  const int wm = (wave & 3) * 32;   // wave M offset (2 tiles)
  const int wn = (wave >> 2) * 64;  // wave N offset (4 tiles)

  // --- staging setup: 2 x 16B chunks per thread per tile -------------------
  // chunk idx: row = idx>>2 (0..127), k8 = (idx&3)*8 (elems)
  const uint32_t lds0 = (uint32_t)(uintptr_t)&smem[0];
  uint32_t voffA[2], voffB[2], ldsA[2], ldsB[2];
#pragma unroll
  for (int i = 0; i < 2; ++i) {
    const int idx = tid + i * 256;
    const int row = idx >> 2;
    const int k8  = (idx & 3) * 8;
    voffA[i] = (uint32_t)(((r0 + row) * HID + k8) * 2);   // bytes into Ah/Al
    voffB[i] = (uint32_t)(((c0 + row) * HID + k8) * 2);   // bytes into Wh/Wl
    const uint32_t tb = (uint32_t)((row * 40 + k8) * 2);  // byte offset in tile
    ldsA[i] = lds0 + tb;                                  // sAh tile base
    ldsB[i] = lds0 + 2 * TILE_BYTES + tb;                 // sBh tile base
  }
  const unsigned long long pAh = (unsigned long long)Ah;
  const unsigned long long pAl = (unsigned long long)Al;
  const unsigned long long pWh = (unsigned long long)Wh;
  const unsigned long long pWl = (unsigned long long)Wl;

  // Issue one k-step's 8 async LDS fills (hi/lo x A/B x 2 chunks), advance k.
  auto issue = [&](int buf) {
    const uint32_t bo = buf ? (uint32_t)BUF_BYTES : 0u;
#pragma unroll
    for (int i = 0; i < 2; ++i) {
      const uint32_t la = ldsA[i] + bo;
      const uint32_t lb = ldsB[i] + bo;
      asm volatile("global_load_async_to_lds_b128 %0, %1, %2"
                   :: "v"(la), "v"(voffA[i]), "s"(pAh) : "memory");
      asm volatile("global_load_async_to_lds_b128 %0, %1, %2"
                   :: "v"(la + (uint32_t)TILE_BYTES), "v"(voffA[i]), "s"(pAl)
                   : "memory");
      asm volatile("global_load_async_to_lds_b128 %0, %1, %2"
                   :: "v"(lb), "v"(voffB[i]), "s"(pWh) : "memory");
      asm volatile("global_load_async_to_lds_b128 %0, %1, %2"
                   :: "v"(lb + (uint32_t)TILE_BYTES), "v"(voffB[i]), "s"(pWl)
                   : "memory");
      voffA[i] += 64;   // next 32-elem k-step, bf16 bytes
      voffB[i] += 64;
    }
  };

  v8f acc[2][4] = {};

  issue(0);   // prologue: k-step 0 into buffer 0

  for (int it = 0; it < HID / 32; ++it) {
    const int buf = it & 1;
    // My fills of `buf` are complete after this wait; the barrier then proves
    // (a) everyone's fills of `buf` are visible and (b) everyone finished
    // reading `buf^1` last step -> safe to read `buf` and to refill `buf^1`.
    asm volatile("s_wait_asynccnt 0x0" ::: "memory");
    __syncthreads();
    if (it + 1 < HID / 32) issue(buf ^ 1);  // overlap next fill with compute

    const __bf16* sAh = smem + (size_t)buf * BUF_ELEMS;
    const __bf16* sAl = sAh + TILE_ELEMS;
    const __bf16* sBh = sAl + TILE_ELEMS;
    const __bf16* sBl = sBh + TILE_ELEMS;

    // A fragments (16x32 bf16 layout): lane holds row m=l16,
    //   k = half*8 + j (j<8), k = 16 + half*8 + (j-8) (j>=8) -> 2 b128 reads.
    v16bf a_hi[2], a_lo[2];
#pragma unroll
    for (int mt = 0; mt < 2; ++mt) {
      const int base = (wm + mt * 16 + l16) * 40 + half * 8;
      {
        v8bf p0 = *(const v8bf*)&sAh[base];
        v8bf p1 = *(const v8bf*)&sAh[base + 16];
        a_hi[mt] = __builtin_shufflevector(p0, p1, 0, 1, 2, 3, 4, 5, 6, 7,
                                           8, 9, 10, 11, 12, 13, 14, 15);
      }
      {
        v8bf p0 = *(const v8bf*)&sAl[base];
        v8bf p1 = *(const v8bf*)&sAl[base + 16];
        a_lo[mt] = __builtin_shufflevector(p0, p1, 0, 1, 2, 3, 4, 5, 6, 7,
                                           8, 9, 10, 11, 12, 13, 14, 15);
      }
    }

    // B fragments (32x16 bf16 layout): lane holds col n=l16,
    //   k = half*8 + j (j<8), k = 16 + half*8 + (j-8) (j>=8)
#pragma unroll
    for (int nt = 0; nt < 4; ++nt) {
      const int base = (wn + nt * 16 + l16) * 40 + half * 8;
      v8bf q0 = *(const v8bf*)&sBh[base];
      v8bf q1 = *(const v8bf*)&sBh[base + 16];
      v16bf b_hi = __builtin_shufflevector(q0, q1, 0, 1, 2, 3, 4, 5, 6, 7,
                                           8, 9, 10, 11, 12, 13, 14, 15);
      v8bf q2 = *(const v8bf*)&sBl[base];
      v8bf q3 = *(const v8bf*)&sBl[base + 16];
      v16bf b_lo = __builtin_shufflevector(q2, q3, 0, 1, 2, 3, 4, 5, 6, 7,
                                           8, 9, 10, 11, 12, 13, 14, 15);
#pragma unroll
      for (int mt = 0; mt < 2; ++mt) {
        acc[mt][nt] = __builtin_amdgcn_wmma_f32_16x16x32_bf16(
            false, a_hi[mt], false, b_hi, (short)0, acc[mt][nt], false, false);
        acc[mt][nt] = __builtin_amdgcn_wmma_f32_16x16x32_bf16(
            false, a_hi[mt], false, b_lo, (short)0, acc[mt][nt], false, false);
        acc[mt][nt] = __builtin_amdgcn_wmma_f32_16x16x32_bf16(
            false, a_lo[mt], false, b_hi, (short)0, acc[mt][nt], false, false);
      }
    }
    // no trailing barrier: next iteration's top barrier covers buffer reuse
  }

  // Epilogue: bias + store. C/D layout: vgpr i -> m = half*8 + i, n = l16.
#pragma unroll
  for (int nt = 0; nt < 4; ++nt) {
    const int n = c0 + wn + nt * 16 + l16;
    const float bv = bias[n];
#pragma unroll
    for (int mt = 0; mt < 2; ++mt) {
#pragma unroll
      for (int i = 0; i < 8; ++i) {
        const int m = r0 + wm + mt * 16 + half * 8 + i;
        C[(size_t)m * NCOL + n] = acc[mt][nt][i] + bv;
      }
    }
  }
}

// ---------------------------------------------------------------------------
// Kernel 2: Vsuf[t,b,nh,d] = sum_{t'>=t} v[t',b,nh,d]
// One lane per (b,nh,d) column (2048 lanes), descending scan over t.
// ---------------------------------------------------------------------------
__global__ __launch_bounds__(256) void suffix_v_kernel(
    const float* __restrict__ qkv, float* __restrict__ vsuf)
{
  const int idx = blockIdx.x * blockDim.x + threadIdx.x;  // 0..2047
  const int d  = idx & 63;
  const int nh = (idx >> 6) & 15;
  const int b  = idx >> 10;
  const float* src = qkv + (size_t)b * NCOL + nh * 192 + 128 + d;  // v block
  float* dst = vsuf + (size_t)b * HID + nh * HDIM + d;
  float acc = 0.0f;
  for (int t = S_LEN - 1; t >= 0; --t) {
    acc += src[(size_t)t * (BATCH * NCOL)];
    dst[(size_t)t * (BATCH * HID)] = acc;
  }
}

// ---------------------------------------------------------------------------
// Kernel 3: per head M[e,d] = scale * sum_t K[t,e] * Vsuf[t,d]   (64x64)
// One block per head (32 heads), LDS-staged t-chunks of 32.
// ---------------------------------------------------------------------------
__global__ __launch_bounds__(256) void kv_head_kernel(
    const float* __restrict__ qkv, const float* __restrict__ vsuf,
    float* __restrict__ Mh)
{
  const int head = blockIdx.x;       // b*16 + nh
  const int b  = head >> 4;
  const int nh = head & 15;
  __shared__ float Kc[32][64];
  __shared__ float Vc[32][64];
  const int tid = threadIdx.x;
  const int e  = tid >> 2;           // 0..63
  const int d0 = (tid & 3) * 16;

  float acc[16];
#pragma unroll
  for (int j = 0; j < 16; ++j) acc[j] = 0.0f;

  for (int t0 = 0; t0 < S_LEN; t0 += 32) {
#pragma unroll
    for (int i = 0; i < 2; ++i) {
      const int idx = tid + i * 256;       // 0..511
      const int tt = idx >> 4;
      const int c4 = (idx & 15) << 2;
      const int r = (t0 + tt) * BATCH + b; // qkv row = s*B + b
      *(float4*)&Kc[tt][c4] = *(const float4*)&qkv[(size_t)r * NCOL + nh * 192 + 64 + c4];
      *(float4*)&Vc[tt][c4] = *(const float4*)&vsuf[(size_t)r * HID + nh * HDIM + c4];
    }
    __syncthreads();
#pragma unroll 8
    for (int tt = 0; tt < 32; ++tt) {
      const float kv = Kc[tt][e];
#pragma unroll
      for (int j = 0; j < 16; ++j) acc[j] += kv * Vc[tt][d0 + j];
    }
    __syncthreads();
  }
#pragma unroll
  for (int j = 0; j < 16; ++j)
    Mh[((size_t)head * 64 + e) * 64 + d0 + j] = acc[j] * kScale;
}

// ---------------------------------------------------------------------------
// Kernel 4: ctx[s,b,nh,d] = sum_e q[s,b,nh,e] * M[head][e,d]
// Grid (S/64, B, NH); block stages M (64x64) and a 64-row Q slab in LDS.
// ---------------------------------------------------------------------------
__global__ __launch_bounds__(256) void out_gemm_kernel(
    const float* __restrict__ qkv, const float* __restrict__ Mh,
    float* __restrict__ out)
{
  const int s0 = blockIdx.x * 64;
  const int b  = blockIdx.y;
  const int nh = blockIdx.z;
  const int head = b * 16 + nh;
  __shared__ float Ms[64][64];
  __shared__ float Qs[64][64];
  const int tid = threadIdx.x;

#pragma unroll
  for (int i = 0; i < 4; ++i) {
    const int idx = tid + i * 256;       // 0..1023
    const int row = idx >> 4;
    const int c4  = (idx & 15) << 2;
    *(float4*)&Ms[row][c4] = *(const float4*)&Mh[((size_t)head * 64 + row) * 64 + c4];
    const int r = (s0 + row) * BATCH + b;
    *(float4*)&Qs[row][c4] = *(const float4*)&qkv[(size_t)r * NCOL + nh * 192 + c4];
  }
  __syncthreads();

  const int sr = tid >> 2;            // local s row 0..63
  const int d0 = (tid & 3) * 16;
  float acc[16];
#pragma unroll
  for (int j = 0; j < 16; ++j) acc[j] = 0.0f;
#pragma unroll 8
  for (int e = 0; e < 64; ++e) {
    const float qv = Qs[sr][e];
#pragma unroll
    for (int j = 0; j < 16; ++j) acc[j] += qv * Ms[e][d0 + j];
  }
  const int r = (s0 + sr) * BATCH + b;
#pragma unroll
  for (int j = 0; j < 16; ++j)
    out[(size_t)r * HID + nh * HDIM + d0 + j] = acc[j];
}

// ---------------------------------------------------------------------------
extern "C" void kernel_launch(void* const* d_in, const int* in_sizes, int n_in,
                              void* d_out, int out_size, void* d_ws, size_t ws_size,
                              hipStream_t stream) {
  (void)in_sizes; (void)n_in; (void)out_size; (void)ws_size;
  const float* hidden = (const float*)d_in[0];  // [S,B,H] fp32
  const float* W      = (const float*)d_in[1];  // [3H,H]  fp32
  const float* bias   = (const float*)d_in[2];  // [3H]    fp32
  float* out = (float*)d_out;                   // [S,B,H] fp32

  // Workspace layout (bytes)
  char* ws = (char*)d_ws;
  float*  qkv  = (float*)ws;                       ws += (size_t)ROWS * NCOL * 4;  // 50.3 MB
  float*  vsuf = (float*)ws;                       ws += (size_t)ROWS * HID * 4;   // 16.8 MB
  float*  Mh   = (float*)ws;                       ws += (size_t)BATCH * NHEAD * HDIM * HDIM * 4;
  __bf16* Ah   = (__bf16*)ws;                      ws += (size_t)ROWS * HID * 2;   // 8.4 MB
  __bf16* Al   = (__bf16*)ws;                      ws += (size_t)ROWS * HID * 2;
  __bf16* Wh   = (__bf16*)ws;                      ws += (size_t)NCOL * HID * 2;   // 6.3 MB
  __bf16* Wl   = (__bf16*)ws;

  const int nA4 = ROWS * HID / 4;   // 1,048,576
  const int nW4 = NCOL * HID / 4;   //   786,432
  split_bf16_kernel<<<dim3(nA4 / 256), 256, 0, stream>>>(hidden, Ah, Al, nA4);
  split_bf16_kernel<<<dim3(nW4 / 256), 256, 0, stream>>>(W, Wh, Wl, nW4);

  qkv_gemm_kernel<<<dim3(ROWS / 128, NCOL / 128), 256, 0, stream>>>(
      Ah, Al, Wh, Wl, bias, qkv);
  suffix_v_kernel<<<dim3(8), 256, 0, stream>>>(qkv, vsuf);
  kv_head_kernel<<<dim3(BATCH * NHEAD), 256, 0, stream>>>(qkv, vsuf, Mh);
  out_gemm_kernel<<<dim3(S_LEN / 64, BATCH, NHEAD), 256, 0, stream>>>(qkv, Mh, out);
}
